// ResidualVectorQuantizer_14834817040989
// MI455X (gfx1250) — matrix operand order repaired and checked by reading, not compiled, not run on hardware
//
#include <hip/hip_runtime.h>

typedef __attribute__((ext_vector_type(8)))  float  v8f;
typedef __attribute__((ext_vector_type(16))) __bf16 v16bf;
typedef __attribute__((ext_vector_type(8)))  __bf16 v8bf;
typedef __attribute__((__vector_size__(16))) int    vi4;

#define NUM_LEVELS 8
#define KCODES 1024
#define DIM 256
#define NTOK 65536
#define BETA 0.25f
#define RVQ_EPS 1e-6f

#if __has_builtin(__builtin_amdgcn_global_load_async_to_lds_b128) && \
    __has_builtin(__builtin_amdgcn_s_wait_asynccnt)
#define HAVE_ASYNC_LDS 1
#else
#define HAVE_ASYNC_LDS 0
#endif

// ---- bf16 helpers (RNE split: x ~= hi + lo) ----
static __device__ __forceinline__ __bf16 f2bf(float x) {
    unsigned u = __float_as_uint(x);
    unsigned r = (u + 0x7FFFu + ((u >> 16) & 1u)) >> 16;
    unsigned short s = (unsigned short)r;
    return __builtin_bit_cast(__bf16, s);
}
static __device__ __forceinline__ float bf2f(__bf16 b) {
    unsigned short s = __builtin_bit_cast(unsigned short, b);
    return __uint_as_float((unsigned)s << 16);
}

// ---------------- init: zero histogram + SSE accumulator ----------------
__global__ void rvq_init(unsigned int* __restrict__ counts, float* __restrict__ sse) {
    int t = threadIdx.x;
    if (t < KCODES) counts[t] = 0u;
    if (t == 0) *sse = 0.0f;
}

// ------- prep: per (level,code) row -> ||c||^2, bf16 hi, bf16 lo -------
__global__ void rvq_prep(const float* __restrict__ cb, float* __restrict__ csq,
                         __bf16* __restrict__ cbhi, __bf16* __restrict__ cblo) {
    int row = blockIdx.x * blockDim.x + threadIdx.x;   // 8192 rows
    if (row < NUM_LEVELS * KCODES) {
        const float* p = cb + (size_t)row * DIM;
        __bf16* ph = cbhi + (size_t)row * DIM;
        __bf16* pl = cblo + (size_t)row * DIM;
        float s = 0.0f;
        #pragma unroll 4
        for (int j = 0; j < DIM; ++j) {
            float v = p[j];
            s += v * v;
            __bf16 h = f2bf(v);
            ph[j] = h;
            pl[j] = f2bf(v - bf2f(h));
        }
        csq[row] = s;
    }
}

// ---------------- main RVQ kernel: 16 tokens / block, 4 waves ----------------
__global__ __launch_bounds__(128) void rvq_main(
    const float* __restrict__ z,
    const float* __restrict__ cb,
    const float* __restrict__ csq,
    const __bf16* __restrict__ cbhi,
    const __bf16* __restrict__ cblo,
    float* __restrict__ out_zq,
    float* __restrict__ out_idx,
    unsigned int* __restrict__ counts,
    float* __restrict__ sse_acc)
{
    __shared__ __align__(16) float  Rs[16 * DIM];    // residual tile (f32)
    __shared__ __align__(16) float  Qs[16 * DIM];    // quantized sum tile
    __shared__ __align__(16) __bf16 Rhi[16 * DIM];   // residual hi (bf16)
    __shared__ __align__(16) __bf16 Rlo[16 * DIM];   // residual lo (bf16)
    __shared__ unsigned long long best[16];          // packed (dist|idx) argmin
    __shared__ float red[16][8][3];                  // partial rc, rr, cc
    __shared__ float kA[16], kB[16];
    __shared__ float blockSse;

    const int tid  = threadIdx.x;
    const int lane = tid & 31;
    const int wave = tid >> 5;
    const int tok0 = blockIdx.x * 16;

    // ---- stage residual tile into LDS (async-to-LDS when available) ----
#if HAVE_ASYNC_LDS
    for (int i = tid; i < 16 * DIM / 4; i += 128) {
        __builtin_amdgcn_global_load_async_to_lds_b128(
            (__attribute__((address_space(1))) vi4*)((const float4*)z + (size_t)tok0 * (DIM / 4) + i),
            (__attribute__((address_space(3))) vi4*)((float4*)Rs + i),
            0, 0);
    }
    __builtin_amdgcn_s_wait_asynccnt(0);
#else
    for (int i = tid; i < 16 * DIM / 4; i += 128) {
        ((float4*)Rs)[i] = ((const float4*)z)[(size_t)tok0 * (DIM / 4) + i];
    }
#endif
    for (int i = tid; i < 16 * DIM / 4; i += 128) {
        float4 zero4 = {0.f, 0.f, 0.f, 0.f};
        ((float4*)Qs)[i] = zero4;
    }
    if (tid == 0) blockSse = 0.0f;
    __syncthreads();
    // bf16 hi/lo decomposition of the residual tile
    for (int i = tid; i < 16 * DIM; i += 128) {
        float v = Rs[i];
        __bf16 h = f2bf(v);
        Rhi[i] = h;
        Rlo[i] = f2bf(v - bf2f(h));
    }
    __syncthreads();

    // WMMA fragment coordinates (ISA 7.12.2, 16-bit layouts, wave32):
    // A 16x32: lane group g supplies K in [8g,8g+8) (VGPR0-3) and [16+8g,16+8g+8) (VGPR4-7), M=lane&15.
// B 32x16: lane group g supplies K in [8g,8g+8) and [16+8g,16+8g+8) (same as A), N=lane&15.
    const int mrow = lane & 15;
    const int g    = lane >> 4;
    const int aoff = g * 8;
    const int boff = g * 8;

    for (int lvl = 0; lvl < NUM_LEVELS; ++lvl) {
        const float*  C   = cb   + (size_t)lvl * KCODES * DIM;
        const __bf16* CH  = cbhi + (size_t)lvl * KCODES * DIM;
        const __bf16* CL  = cblo + (size_t)lvl * KCODES * DIM;
        const float*  CS  = csq  + lvl * KCODES;

        if (tid < 16) best[tid] = ~0ull;
        __syncthreads();

        // ---- distance GEMM (bf16 hi/lo split) + argmin, 64 codes per pass ----
        for (int chunk = 0; chunk < 16; ++chunk) {
            const int n = chunk * 64 + wave * 16 + mrow;   // this lane's code column
            const __bf16* Bh = CH + (size_t)n * DIM;
            const __bf16* Bl = CL + (size_t)n * DIM;
            __builtin_prefetch(CH + (size_t)((n + 64) & (KCODES - 1)) * DIM, 0, 0);

            const __bf16* Arh = &Rhi[mrow * DIM];
            const __bf16* Arl = &Rlo[mrow * DIM];

            v8f acc = {};
            #pragma unroll
            for (int k0 = 0; k0 < DIM; k0 += 32) {
                v8bf ah0 = *(const v8bf*)(Arh + k0 + aoff);
                v8bf ah1 = *(const v8bf*)(Arh + k0 + 16 + aoff);
                v8bf al0 = *(const v8bf*)(Arl + k0 + aoff);
                v8bf al1 = *(const v8bf*)(Arl + k0 + 16 + aoff);
                v16bf Ah = __builtin_shufflevector(ah0, ah1, 0,1,2,3,4,5,6,7,8,9,10,11,12,13,14,15);
                v16bf Al = __builtin_shufflevector(al0, al1, 0,1,2,3,4,5,6,7,8,9,10,11,12,13,14,15);
                v8bf bh0 = *(const v8bf*)(Bh + k0 + boff);
                v8bf bh1 = *(const v8bf*)(Bh + k0 + boff + 16);
                v8bf bl0 = *(const v8bf*)(Bl + k0 + boff);
                v8bf bl1 = *(const v8bf*)(Bl + k0 + boff + 16);
                v16bf Bhv = __builtin_shufflevector(bh0, bh1, 0,1,2,3,4,5,6,7,8,9,10,11,12,13,14,15);
                v16bf Blv = __builtin_shufflevector(bl0, bl1, 0,1,2,3,4,5,6,7,8,9,10,11,12,13,14,15);
                // r.c ~= hi.hi + hi.lo + lo.hi  (lo.lo negligible)
                acc = __builtin_amdgcn_wmma_f32_16x16x32_bf16(false, Ah, false, Bhv, (short)0, acc, false, false);
                acc = __builtin_amdgcn_wmma_f32_16x16x32_bf16(false, Ah, false, Blv, (short)0, acc, false, false);
                acc = __builtin_amdgcn_wmma_f32_16x16x32_bf16(false, Al, false, Bhv, (short)0, acc, false, false);
            }

            const float cs = CS[n];
            #pragma unroll
            for (int i = 0; i < 8; ++i) {
                // C/D layout: VGPR i holds M=i (lanes 0-15) or M=i+8 (lanes 16-31); N=lane&15.
                int   mm   = i + (g << 3);
                float dist = cs - 2.0f * acc[i];
                unsigned ub = __float_as_uint(dist);
                ub = (ub & 0x80000000u) ? ~ub : (ub | 0x80000000u);   // order-preserving map
                unsigned long long key =
                    ((unsigned long long)ub << 32) | (unsigned)n;      // low bits: index (tie -> smallest)
                atomicMin(&best[mm], key);                             // ds_min_u64
            }
        }
        __syncthreads();

        // ---- gather winner, losses, rotation-trick update (8 threads / token) ----
        const int tok = tid >> 3;   // 0..15
        const int sub = tid & 7;    // 0..7, 32 dims each
        const int idx = (int)(unsigned)(best[tok] & 0xFFFFFFFFu);

        const float4* Cw = (const float4*)(C + (size_t)idx * DIM + sub * 32);
        float4 creg[8];
        float prc = 0.f, prr = 0.f, pcc = 0.f;
        #pragma unroll
        for (int j = 0; j < 8; ++j) {
            float4 c4 = Cw[j];
            creg[j] = c4;
            float4 r4 = *(const float4*)&Rs[tok * DIM + sub * 32 + j * 4];
            prc += r4.x * c4.x + r4.y * c4.y + r4.z * c4.z + r4.w * c4.w;
            prr += r4.x * r4.x + r4.y * r4.y + r4.z * r4.z + r4.w * r4.w;
            pcc += c4.x * c4.x + c4.y * c4.y + c4.z * c4.z + c4.w * c4.w;
        }
        red[tok][sub][0] = prc;
        red[tok][sub][1] = prr;
        red[tok][sub][2] = pcc;
        __syncthreads();

        if (sub == 0) {
            float rc = 0.f, rr = 0.f, cc = 0.f;
            #pragma unroll
            for (int j = 0; j < 8; ++j) {
                rc += red[tok][j][0];
                rr += red[tok][j][1];
                cc += red[tok][j][2];
            }
            // rotation trick forward value: r_new = ka*r + kb*c
            float nr = fmaxf(sqrtf(rr), RVQ_EPS);
            float nc = fmaxf(sqrtf(cc), RVQ_EPS);
            float uu = rr / (nr * nr);
            float qq = cc / (nc * nc);
            float uq = rc / (nr * nc);
            float ns = fmaxf(sqrtf(uu + 2.0f * uq + qq), RVQ_EPS);
            float rw = (rr / nr + rc / nc) / ns;        // r . w
            float ru = rr / nr;                          // r . u
            float A  = 2.0f * rw / ns;
            float B  = 2.0f * ru / nc;
            kA[tok] = A / nr;
            kB[tok] = A / nc - B;
            atomicAdd(&blockSse, cc - 2.0f * rc + rr);   // ||c - r||^2
            atomicAdd(&counts[idx], 1u);
            out_idx[(size_t)(tok0 + tok) * NUM_LEVELS + lvl] = (float)idx;
        }
        __syncthreads();

        const float ka = kA[tok];
        const float kb = kB[tok];
        #pragma unroll
        for (int j = 0; j < 8; ++j) {
            const int o = tok * DIM + sub * 32 + j * 4;
            float* rp = &Rs[o];
            float* qp = &Qs[o];
            float4 r4 = *(float4*)rp;
            float4 q4 = *(float4*)qp;
            r4.x = ka * r4.x + kb * creg[j].x;  q4.x += creg[j].x;
            r4.y = ka * r4.y + kb * creg[j].y;  q4.y += creg[j].y;
            r4.z = ka * r4.z + kb * creg[j].z;  q4.z += creg[j].z;
            r4.w = ka * r4.w + kb * creg[j].w;  q4.w += creg[j].w;
            *(float4*)rp = r4;
            *(float4*)qp = q4;
            // refresh bf16 hi/lo residual for next level's WMMA
            __bf16 h;
            h = f2bf(r4.x); Rhi[o + 0] = h; Rlo[o + 0] = f2bf(r4.x - bf2f(h));
            h = f2bf(r4.y); Rhi[o + 1] = h; Rlo[o + 1] = f2bf(r4.y - bf2f(h));
            h = f2bf(r4.z); Rhi[o + 2] = h; Rlo[o + 2] = f2bf(r4.z - bf2f(h));
            h = f2bf(r4.w); Rhi[o + 3] = h; Rlo[o + 3] = f2bf(r4.w - bf2f(h));
        }
        __syncthreads();
    }

    // straight-through forward value: z_q = quantized_sum
    for (int i = tid; i < 16 * DIM / 4; i += 128) {
        ((float4*)out_zq)[(size_t)tok0 * (DIM / 4) + i] = ((const float4*)Qs)[i];
    }
    if (tid == 0) atomicAdd(sse_acc, blockSse);
}

// ---------------- finalize: loss scalars + perplexity ----------------
__global__ void rvq_final(const unsigned int* __restrict__ counts,
                          const float* __restrict__ sse,
                          float* __restrict__ out_scalars)
{
    __shared__ float partial[256];
    float h = 0.0f;
    for (int i = threadIdx.x; i < KCODES; i += 256) {
        float p = (float)counts[i] * (1.0f / (float)(NTOK * NUM_LEVELS));
        if (p > 0.0f) h -= p * logf(p);
    }
    partial[threadIdx.x] = h;
    __syncthreads();
    for (int s = 128; s > 0; s >>= 1) {
        if (threadIdx.x < s) partial[threadIdx.x] += partial[threadIdx.x + s];
        __syncthreads();
    }
    if (threadIdx.x == 0) {
        float cl = *sse * (1.0f / (float)((size_t)NTOK * DIM));  // = codebook = commitment
        out_scalars[0] = cl + BETA * cl;   // loss
        out_scalars[1] = cl;               // total_codebook_loss
        out_scalars[2] = cl;               // total_commitment_loss
        out_scalars[3] = expf(partial[0]); // perplexity
    }
}

extern "C" void kernel_launch(void* const* d_in, const int* in_sizes, int n_in,
                              void* d_out, int out_size, void* d_ws, size_t ws_size,
                              hipStream_t stream) {
    const float* z  = (const float*)d_in[0];
    const float* cb = (const float*)d_in[1];
    float* out = (float*)d_out;

    unsigned int* counts = (unsigned int*)d_ws;                       // 1024 * 4B
    float*  sse  = (float*)((char*)d_ws + 4096);                      // 4B
    float*  csq  = (float*)((char*)d_ws + 8192);                      // 8192 * 4B
    __bf16* cbhi = (__bf16*)((char*)d_ws + 65536);                    // 4 MB
    __bf16* cblo = (__bf16*)((char*)d_ws + 65536 + (size_t)NUM_LEVELS * KCODES * DIM * 2);

    float* out_zq      = out;                                         // 65536*256
    float* out_idx     = out + (size_t)NTOK * DIM;                    // 65536*8
    float* out_scalars = out + (size_t)NTOK * DIM + (size_t)NTOK * NUM_LEVELS;

    rvq_init<<<1, 1024, 0, stream>>>(counts, sse);
    rvq_prep<<<(NUM_LEVELS * KCODES + 255) / 256, 256, 0, stream>>>(cb, csq, cbhi, cblo);
    rvq_main<<<NTOK / 16, 128, 0, stream>>>(z, cb, csq, cbhi, cblo, out_zq, out_idx, counts, sse);
    rvq_final<<<1, 256, 0, stream>>>(counts, sse, out_scalars);
}
